// KmeansQuantizer_52673478918654
// MI455X (gfx1250) — matrix operand order repaired and checked by reading, not compiled, not run on hardware
//
#include <hip/hip_runtime.h>
#include <hip/hip_bf16.h>
#include <stdint.h>

// MI455X / gfx1250, wave32. D = AxB+C via v_wmma_f32_16x16x32_bf16.
//
// argmin_c ||x-c||^2 == argmin_c (c^2 - 2 x.c)  -> GEMM (split-bf16, 3 WMMA
// passes: hi*hi + hi*lo + lo*hi, f32 accumulate) + per-row argmin.
//
// ws layout: [0,4KB) c2[1024] f32 ; [4KB, +1.5MB) Bhi swizzled bf16 ;
//            [+1.5MB, +3MB) Blo swizzled bf16.  Needs ws_size >= 3,149,824 B.

typedef __attribute__((ext_vector_type(16))) __bf16 v16bf;
typedef __attribute__((ext_vector_type(8)))  float  v8f;

#define BB 16
#define TT 4096
#define DD 768
#define CC 1024
#define M_TOTAL (BB * TT)   // 65536 rows
#define KC (DD / 32)        // 24 k-steps of 32
#define NC (CC / 16)        // 64 col chunks of 16

// ---------------- prep: c2[c] = sum_d clusters[c][d]^2 ----------------
__global__ void Kmq_c2_kernel(const float* __restrict__ clusters,
                              float* __restrict__ c2) {
    int c = blockIdx.x * blockDim.x + threadIdx.x;
    if (c >= CC) return;
    const float* p = clusters + (size_t)c * DD;
    float s = 0.f;
    for (int d = 0; d < DD; d += 4) {
        float4 v = *(const float4*)(p + d);
        s = fmaf(v.x, v.x, s); s = fmaf(v.y, v.y, s);
        s = fmaf(v.z, v.z, s); s = fmaf(v.w, v.w, s);
    }
    c2[c] = s;
}

// ---- prep: swizzle clusters into WMMA-B operand layout, split bf16 ----
// B operand (32x16) in this code's contiguous k order: lane n (0-15) holds col N=n, K=0..15 packed
// 2-per-VGPR; lanes 16-31 hold same cols, K=16..31.  We store, per chunk
// (nc,kc) and lane, the 16 bf16 elements contiguously (32B per lane).
__global__ void Kmq_swz_kernel(const float* __restrict__ clusters,
                               __bf16* __restrict__ bhi,
                               __bf16* __restrict__ blo) {
    int t = blockIdx.x * blockDim.x + threadIdx.x;   // nc*KC*32 + kc*32 + lane
    if (t >= NC * KC * 32) return;
    int lane = t & 31;
    int kc   = (t >> 5) % KC;
    int nc   = t / (32 * KC);
    int col  = nc * 16 + (lane & 15);
    int kbase = kc * 32 + (lane >> 4) * 16;
    const float* src = clusters + (size_t)col * DD + kbase;
    __bf16* dh = bhi + (size_t)t * 16;
    __bf16* dl = blo + (size_t)t * 16;
#pragma unroll
    for (int e = 0; e < 16; ++e) {
        float x = src[e];
        __bf16 h = (__bf16)x;
        dh[e] = h;
        dl[e] = (__bf16)(x - (float)h);
    }
}

// ------------------------- main assignment kernel -------------------------
// 256 threads = 8 waves; wave w handles 16 rows (blockIdx.x*128 + w*16).
// N loop: 8 outer chunks of 128 cols (8 accumulators of v8f per wave).
// K loop: 24 steps of 32; B tile (8 col-chunks, hi+lo = 16KB) double-buffered
// in LDS: single barrier per step, tile s+1 staged while tile s feeds WMMAs.
__global__ __launch_bounds__(256) void Kmq_assign_kernel(
    const float* __restrict__ feat,
    const __bf16* __restrict__ bhi,
    const __bf16* __restrict__ blo,
    const float* __restrict__ c2,
    float* __restrict__ out) {
    __shared__ __align__(16) __bf16 lds_hi[2][8 * 32 * 16];  // 2 x 8 KB
    __shared__ __align__(16) __bf16 lds_lo[2][8 * 32 * 16];  // 2 x 8 KB

    const int tid  = threadIdx.x;
    const int lane = tid & 31;
    const int wave = tid >> 5;       // 0..7
    const int half = lane >> 4;      // 0/1 (K-half for A, row-half for C/D)
    const int l15  = lane & 15;
    const int rowBase = blockIdx.x * 128 + wave * 16;
    const float* myFeat = feat + (size_t)(rowBase + l15) * DD;

    // Per-thread staging slot: 32B of hi + 32B of lo per (no,kc) tile.
    auto stage = [&](int sno, int skc, int buf) {
        size_t so = ((size_t)((sno * 8 + wave) * KC + skc) * 32 + lane) * 16;
        const uint4* sh = (const uint4*)(bhi + so);
        const uint4* sl = (const uint4*)(blo + so);
        uint4* dh = (uint4*)(lds_hi[buf] + (wave * 32 + lane) * 16);
        uint4* dl = (uint4*)(lds_lo[buf] + (wave * 32 + lane) * 16);
        dh[0] = sh[0]; dh[1] = sh[1];
        dl[0] = sl[0]; dl[1] = sl[1];
    };

    float bestv[8], besti[8];
#pragma unroll
    for (int r = 0; r < 8; ++r) { bestv[r] = 3.4e38f; besti[r] = 0.f; }

    stage(0, 0, 0);   // prime buffer 0 with tile (no=0,kc=0)
    int buf = 0;

    for (int no = 0; no < 8; ++no) {
        v8f acc[8] = {};
        for (int kc = 0; kc < KC; ++kc) {
            // ---- load A (fp32) early and split to bf16 hi/lo; latency
            // overlaps the barrier wait.  A operand: lane half h holds
            // K = 16h+{0..15}.
            const float* ap = myFeat + kc * 32 + half * 16;
            float4 f0 = *(const float4*)(ap + 0);
            float4 f1 = *(const float4*)(ap + 4);
            float4 f2 = *(const float4*)(ap + 8);
            float4 f3 = *(const float4*)(ap + 12);
            float xs[16] = {f0.x, f0.y, f0.z, f0.w, f1.x, f1.y, f1.z, f1.w,
                            f2.x, f2.y, f2.z, f2.w, f3.x, f3.y, f3.z, f3.w};
            v16bf ahi, alo;
#pragma unroll
            for (int e = 0; e < 16; ++e) {
                float x = xs[e];
                __bf16 h = (__bf16)x;
                ahi[e] = h;
                alo[e] = (__bf16)(x - (float)h);
            }

            __syncthreads();   // tile `buf` staged; buf^1 consumers done

            // ---- prefetch next tile into the other buffer
            int nkc = kc + 1, nno = no;
            if (nkc == KC) { nkc = 0; nno = no + 1; }
            if (nno < 8) stage(nno, nkc, buf ^ 1);

            // ---- 8 col-chunks x 3 split-precision WMMAs from tile `buf`
#pragma unroll
            for (int ni = 0; ni < 8; ++ni) {
                v16bf bh = *(const v16bf*)(lds_hi[buf] + (ni * 32 + lane) * 16);
                v16bf bl = *(const v16bf*)(lds_lo[buf] + (ni * 32 + lane) * 16);
                acc[ni] = __builtin_amdgcn_wmma_f32_16x16x32_bf16(
                    false, ahi, false, bh, (short)0, acc[ni], false, false);
                acc[ni] = __builtin_amdgcn_wmma_f32_16x16x32_bf16(
                    false, ahi, false, bl, (short)0, acc[ni], false, false);
                acc[ni] = __builtin_amdgcn_wmma_f32_16x16x32_bf16(
                    false, alo, false, bh, (short)0, acc[ni], false, false);
            }
            buf ^= 1;
        }
        // ---- fold scores into running per-lane argmin.
        // C/D layout: VGPR r, lanes 0-15 -> row r, lanes 16-31 -> row r+8;
        // col = l15 (+chunk offsets). score = c2[col] - 2*dot.
#pragma unroll
        for (int ni = 0; ni < 8; ++ni) {
            int col = no * 128 + ni * 16 + l15;
            float cc = c2[col];
            float colf = (float)col;
#pragma unroll
            for (int r = 0; r < 8; ++r) {
                float s = fmaf(-2.0f, acc[ni][r], cc);
                if (s < bestv[r]) { bestv[r] = s; besti[r] = colf; }
            }
        }
    }
    // ---- final argmin across the 16 col-slots (16-lane groups)
#pragma unroll
    for (int r = 0; r < 8; ++r) {
        float v = bestv[r], i = besti[r];
#pragma unroll
        for (int off = 1; off < 16; off <<= 1) {
            float ov = __shfl_xor(v, off, 16);
            float oi = __shfl_xor(i, off, 16);
            if (ov < v || (ov == v && oi < i)) { v = ov; i = oi; }
        }
        if (l15 == 0) out[rowBase + r + half * 8] = i;
    }
}

extern "C" void kernel_launch(void* const* d_in, const int* in_sizes, int n_in,
                              void* d_out, int out_size, void* d_ws, size_t ws_size,
                              hipStream_t stream) {
    const float* feat     = (const float*)d_in[0];   // [16,4096,768] f32
    const float* clusters = (const float*)d_in[1];   // [1024,768] f32
    float* out = (float*)d_out;                      // [16,4096] indices-as-f32

    uint8_t* ws = (uint8_t*)d_ws;
    float*  c2  = (float*)ws;                                       // 4 KB
    __bf16* bhi = (__bf16*)(ws + 4096);                             // 1.5 MB
    __bf16* blo = (__bf16*)(ws + 4096 + (size_t)CC * DD * 2);       // 1.5 MB
    (void)in_sizes; (void)n_in; (void)out_size; (void)ws_size;

    Kmq_c2_kernel<<<(CC + 255) / 256, 256, 0, stream>>>(clusters, c2);
    Kmq_swz_kernel<<<(NC * KC * 32 + 255) / 256, 256, 0, stream>>>(clusters, bhi, blo);
    Kmq_assign_kernel<<<M_TOTAL / 128, 256, 0, stream>>>(feat, bhi, blo, c2, out);
}
